// MolGraphTransformer_3813930959493
// MI455X (gfx1250) — matrix operand order repaired and checked by reading, not compiled, not run on hardware
//
#include <hip/hip_runtime.h>
#include <math.h>

#define NNODE  20000
#define NEDGE  320000
#define HIDDEN 128
#define NHEAD  4
#define HDIM   32
#define NLAYER 4
#define NGRAPH 1024
#define NOUT   256

typedef _Float16 f16;
typedef _Float16 v16h __attribute__((ext_vector_type(16)));
typedef _Float16 v8h  __attribute__((ext_vector_type(8)));
typedef float    v8f  __attribute__((ext_vector_type(8)));

__constant__ int c_atom_off[9] = {0,120,130,142,154,164,170,180,182};
__constant__ int c_bond_off[3] = {0,20,30};

// ---------------------------------------------------------------- utilities
__global__ void k_fill(float* p, float v, int n) {
  int i = blockIdx.x * blockDim.x + threadIdx.x;
  if (i < n) p[i] = v;
}

// Pack W[K,Nc] (f32 row-major) into WMMA B-fragment order (f16):
//   Bp[((ntile*(K/32)+kb)*32+lane)*16 + j] = W[(kb*32 + (lane>>4)*16 + j)*Nc + ntile*16+(lane&15)]
// so each lane's 16-half fragment is one contiguous 32-byte vector.
__global__ void k_pack_b(const float* __restrict__ W, f16* __restrict__ Bp, int K, int Nc) {
  int i = blockIdx.x * blockDim.x + threadIdx.x;
  if (i >= K * Nc) return;
  int j     = i & 15;
  int lane  = (i >> 4) & 31;
  int t     = i >> 9;
  int kbs   = K >> 5;
  int kb    = t % kbs;
  int ntile = t / kbs;
  int n     = ntile * 16 + (lane & 15);
  int kIdx  = kb * 32 + (lane >> 4) * 16 + j;
  Bp[i] = (f16)W[(size_t)kIdx * Nc + n];
}

// ------------------------------------------------------------- embeddings
__global__ void k_atom_encode(const float* __restrict__ tab, const int* __restrict__ x,
                              float* __restrict__ hf, f16* __restrict__ hh) {
  int n = blockIdx.x, d = threadIdx.x;
  float acc = 0.f;
#pragma unroll
  for (int f = 0; f < 9; ++f)
    acc += tab[(size_t)(x[n * 9 + f] + c_atom_off[f]) * HIDDEN + d];
  hf[(size_t)n * HIDDEN + d] = acc;
  hh[(size_t)n * HIDDEN + d] = (f16)acc;
}

__global__ void k_bond_encode(const float* __restrict__ tab, const int* __restrict__ ea,
                              f16* __restrict__ eh) {
  int e = blockIdx.x, d = threadIdx.x;
  float acc = 0.f;
#pragma unroll
  for (int f = 0; f < 3; ++f)
    acc += tab[(size_t)(ea[e * 3 + f] + c_bond_off[f]) * HIDDEN + d];
  eh[(size_t)e * HIDDEN + d] = (f16)acc;
}

// ------------------------------------------------------------- WMMA GEMM
// C[M,Nc] = A[M,K](f16 row-major) @ W(pre-packed f16 fragments) (+bias)(relu?)
// Block = 8 waves, all sharing one pair of 16-wide N tiles. The 8KB of packed
// B fragments for the block is staged into LDS once with
// global_load_async_to_lds_b128 (ASYNCcnt) and read back as ds_load_b128.
// Each wave computes a 16x32 output tile: K unrolls to 2*(K/32) back-to-back
// v_wmma_f32_16x16x32_f16 ops. Requires M%16==0, Nc%32==0 (all call sites).
template <int K, bool BIAS, bool RELU, bool OUTF, bool OUTH>
__global__ void k_gemm_wmma(const f16* __restrict__ A, const f16* __restrict__ Bp,
                            const float* __restrict__ bias, float* __restrict__ Cf,
                            f16* __restrict__ Ch, int M, int Nc) {
  constexpr int KB = K / 32;
  __shared__ f16 sB[2 * KB * 512];            // 2 N-tiles of fragments (8KB for K=128)

  int wave   = threadIdx.x >> 5;
  int lane   = threadIdx.x & 31;
  int mtile  = blockIdx.x * 8 + wave;
  int ntile2 = blockIdx.y;                    // pair of 16-wide N tiles
  bool valid = (mtile * 16 < M);              // wave-uniform
  int lan15  = lane & 15;
  int hi     = lane >> 4;

  // ---- async stage of packed B fragments into LDS (512 x 16B chunks) ----
  {
    constexpr int CHUNKS = (2 * KB * 512 * 2) / 16;   // 512 for K=128
    const char* gsrc = (const char*)(Bp + (size_t)(2 * ntile2) * KB * 512);
#pragma unroll
    for (int c = threadIdx.x; c < CHUNKS; c += 256) {
      unsigned int ldsOff = (unsigned int)(c * 16);   // sB is the only LDS object
      unsigned long long ga = (unsigned long long)(gsrc + (size_t)c * 16);
      asm volatile("global_load_async_to_lds_b128 %0, %1, off"
                   :: "v"(ldsOff), "v"(ga) : "memory");
    }
    asm volatile("s_wait_asynccnt 0x0" ::: "memory");
  }
  __syncthreads();

  v8f acc0 = {}, acc1 = {};
  if (valid) {
    const f16* ap  = A + (size_t)(mtile * 16 + lan15) * K + hi * 16;
    const f16* sb0 = sB + lane * 16;
    const f16* sb1 = sB + KB * 512 + lane * 16;
#pragma unroll
    for (int kb = 0; kb < KB; ++kb) {
      v8h alo = *(const v8h*)(ap + kb * 32);
      v8h ahi = *(const v8h*)(ap + kb * 32 + 8);
      v16h av = __builtin_shufflevector(alo, ahi, 0, 1, 2, 3, 4, 5, 6, 7,
                                        8, 9, 10, 11, 12, 13, 14, 15);
      v16h b0 = *(const v16h*)(sb0 + kb * 512);
      v16h b1 = *(const v16h*)(sb1 + kb * 512);
      acc0 = __builtin_amdgcn_wmma_f32_16x16x32_f16(false, av, false, b0,
                                                    (short)0, acc0, false, false);
      acc1 = __builtin_amdgcn_wmma_f32_16x16x32_f16(false, av, false, b1,
                                                    (short)0, acc1, false, false);
    }

    int n0    = 2 * ntile2 * 16 + lan15;
    int n1    = n0 + 16;
    int mbase = mtile * 16 + hi * 8;
    float bb0 = BIAS ? bias[n0] : 0.f;
    float bb1 = BIAS ? bias[n1] : 0.f;
#pragma unroll
    for (int r = 0; r < 8; ++r) {
      float v0 = acc0[r] + bb0, v1 = acc1[r] + bb1;
      if (RELU) { v0 = fmaxf(v0, 0.f); v1 = fmaxf(v1, 0.f); }
      if (OUTF) {
        Cf[(size_t)(mbase + r) * Nc + n0] = v0;
        Cf[(size_t)(mbase + r) * Nc + n1] = v1;
      }
      if (OUTH) {
        Ch[(size_t)(mbase + r) * Nc + n0] = (f16)v0;
        Ch[(size_t)(mbase + r) * Nc + n1] = (f16)v1;
      }
    }
  }
}

template <bool BIAS, bool RELU, bool OUTF, bool OUTH>
static inline void launch_gemm(const f16* A, const f16* Bp, const float* bias,
                               float* Cf, f16* Ch, int M, int Nc, hipStream_t stream) {
  dim3 grid((M / 16 + 7) / 8, Nc / 32);
  k_gemm_wmma<HIDDEN, BIAS, RELU, OUTF, OUTH>
      <<<grid, 256, 0, stream>>>(A, Bp, bias, Cf, Ch, M, Nc);
}

// ------------------------------------------------------------- attention
__device__ inline void atomicMaxFloat(float* addr, float val) {
  int* ia = (int*)addr;
  int old = __float_as_int(*addr);
  while (__int_as_float(old) < val) {
    int assumed = old;
    old = atomicCAS(ia, assumed, __float_as_int(val));
    if (old == assumed) break;
  }
}

__global__ void k_edge_logits(const int* __restrict__ ei, const float* __restrict__ q,
                              const float* __restrict__ k, const f16* __restrict__ ee,
                              float* __restrict__ logits, float* __restrict__ nmax) {
  int e = blockIdx.x * 8 + (threadIdx.x >> 5);
  if (e >= NEDGE) return;
  int lane = threadIdx.x & 31;
  int s = ei[e], d = ei[NEDGE + e];
#pragma unroll
  for (int h = 0; h < NHEAD; ++h) {
    int idx = h * HDIM + lane;
    float kj = k[(size_t)s * HIDDEN + idx] + (float)ee[(size_t)e * HIDDEN + idx];
    float p  = q[(size_t)d * HIDDEN + idx] * kj;
#pragma unroll
    for (int off = 16; off > 0; off >>= 1) p += __shfl_xor(p, off, 32);
    if (lane == 0) {
      float lgt = p * 0.17677669529663687f;  // 1/sqrt(32)
      logits[(size_t)e * NHEAD + h] = lgt;
      atomicMaxFloat(&nmax[(size_t)d * NHEAD + h], lgt);
    }
  }
}

__global__ void k_edge_exp(const int* __restrict__ ei, float* __restrict__ logits,
                           const float* __restrict__ nmax, float* __restrict__ nsum) {
  int i = blockIdx.x * blockDim.x + threadIdx.x;
  if (i >= NEDGE * NHEAD) return;
  int e = i >> 2, h = i & 3;
  int d = ei[NEDGE + e];
  float p = expf(logits[i] - nmax[(size_t)d * NHEAD + h]);
  logits[i] = p;
  atomicAdd(&nsum[(size_t)d * NHEAD + h], p);
}

__global__ void k_edge_accum(const int* __restrict__ ei, const float* __restrict__ logits,
                             const float* __restrict__ nsum, const float* __restrict__ v,
                             const f16* __restrict__ ee, float* __restrict__ msg) {
  int e = blockIdx.x * 8 + (threadIdx.x >> 5);
  if (e >= NEDGE) return;
  int lane = threadIdx.x & 31;
  int s = ei[e], d = ei[NEDGE + e];
#pragma unroll
  for (int h = 0; h < NHEAD; ++h) {
    float alpha = logits[(size_t)e * NHEAD + h] /
                  fmaxf(nsum[(size_t)d * NHEAD + h], 1e-16f);
    int idx = h * HDIM + lane;
    float vj = v[(size_t)s * HIDDEN + idx] + (float)ee[(size_t)e * HIDDEN + idx];
    atomicAdd(&msg[(size_t)d * HIDDEN + idx], alpha * vj);
  }
}

__global__ void k_node_update(const float* __restrict__ msg, const float* __restrict__ skip,
                              float* __restrict__ hf, f16* __restrict__ hh) {
  int i = blockIdx.x * blockDim.x + threadIdx.x;
  if (i >= NNODE * HIDDEN) return;
  float vv = fmaxf(msg[i] + skip[i], 0.f);
  hf[i] = vv;
  hh[i] = (f16)vv;
}

// ------------------------------------------------------------- pooling / head
__global__ void k_pool(const int* __restrict__ batch, const float* __restrict__ hf,
                       float* __restrict__ g, float* __restrict__ cnt) {
  int n = blockIdx.x, d = threadIdx.x;
  int b = batch[n];
  atomicAdd(&g[(size_t)b * HIDDEN + d], hf[(size_t)n * HIDDEN + d]);
  if (d == 0) atomicAdd(&cnt[b], 1.f);
}

__global__ void k_gdiv(const float* __restrict__ g, const float* __restrict__ cnt,
                       f16* __restrict__ g16) {
  int b = blockIdx.x, d = threadIdx.x;
  g16[(size_t)b * HIDDEN + d] = (f16)(g[(size_t)b * HIDDEN + d] / fmaxf(cnt[b], 1.f));
}

__global__ void k_l2norm(float* __restrict__ out) {
  __shared__ float red[NOUT];
  int g = blockIdx.x, t = threadIdx.x;
  float v = out[(size_t)g * NOUT + t];
  red[t] = v * v;
  __syncthreads();
  for (int s = NOUT / 2; s > 0; s >>= 1) {
    if (t < s) red[t] += red[t + s];
    __syncthreads();
  }
  float norm = sqrtf(red[0]);
  out[(size_t)g * NOUT + t] = v / fmaxf(norm, 1e-12f);
}

// ------------------------------------------------------------------ driver
extern "C" void kernel_launch(void* const* d_in, const int* in_sizes, int n_in,
                              void* d_out, int out_size, void* d_ws, size_t ws_size,
                              hipStream_t stream) {
  (void)in_sizes; (void)n_in; (void)out_size; (void)ws_size;

  const float* atom_table = (const float*)d_in[0];
  const float* bond_table = (const float*)d_in[1];
  const float* Wq  = (const float*)d_in[2];
  const float* bq  = (const float*)d_in[3];
  const float* Wk  = (const float*)d_in[4];
  const float* bk  = (const float*)d_in[5];
  const float* Wv  = (const float*)d_in[6];
  const float* bvp = (const float*)d_in[7];
  const float* We  = (const float*)d_in[8];
  const float* Wsk = (const float*)d_in[9];
  const float* bsp = (const float*)d_in[10];
  const float* W1  = (const float*)d_in[11];
  const float* b1  = (const float*)d_in[12];
  const float* W2  = (const float*)d_in[13];
  const float* b2  = (const float*)d_in[14];
  const int* x          = (const int*)d_in[15];
  const int* edge_attr  = (const int*)d_in[16];
  const int* edge_index = (const int*)d_in[17];
  const int* batch      = (const int*)d_in[18];
  float* out = (float*)d_out;

  char* base = (char*)d_ws;
  size_t off = 0;
  auto alloc = [&](size_t bytes) -> char* {
    char* p = base + off;
    off += (bytes + 255) & ~(size_t)255;
    return p;
  };
  float* hf   = (float*)alloc((size_t)NNODE * HIDDEN * 4);
  f16*   hh   = (f16*)  alloc((size_t)NNODE * HIDDEN * 2);
  f16*   eh   = (f16*)  alloc((size_t)NEDGE * HIDDEN * 2);
  f16*   ee   = (f16*)  alloc((size_t)NEDGE * HIDDEN * 2);
  float* qf   = (float*)alloc((size_t)NNODE * HIDDEN * 4);
  float* kf   = (float*)alloc((size_t)NNODE * HIDDEN * 4);
  float* vf   = (float*)alloc((size_t)NNODE * HIDDEN * 4);
  float* skf  = (float*)alloc((size_t)NNODE * HIDDEN * 4);
  float* msg  = (float*)alloc((size_t)NNODE * HIDDEN * 4);
  float* lg   = (float*)alloc((size_t)NEDGE * NHEAD * 4);
  float* nmax = (float*)alloc((size_t)NNODE * NHEAD * 4);
  float* nsum = (float*)alloc((size_t)NNODE * NHEAD * 4);
  float* gacc = (float*)alloc((size_t)NGRAPH * HIDDEN * 4);
  float* cnt  = (float*)alloc((size_t)NGRAPH * 4);
  f16*   g16  = (f16*)  alloc((size_t)NGRAPH * HIDDEN * 2);
  f16*   g116 = (f16*)  alloc((size_t)NGRAPH * HIDDEN * 2);
  const size_t WSZ = (size_t)HIDDEN * HIDDEN;
  f16* wq16 = (f16*)alloc(NLAYER * WSZ * 2);
  f16* wk16 = (f16*)alloc(NLAYER * WSZ * 2);
  f16* wv16 = (f16*)alloc(NLAYER * WSZ * 2);
  f16* we16 = (f16*)alloc(NLAYER * WSZ * 2);
  f16* ws16 = (f16*)alloc(NLAYER * WSZ * 2);
  f16* w116 = (f16*)alloc(WSZ * 2);
  f16* w216 = (f16*)alloc((size_t)HIDDEN * NOUT * 2);

  auto packB = [&](const float* W, f16* Bp, int K, int Nc) {
    int n = K * Nc;
    k_pack_b<<<(n + 255) / 256, 256, 0, stream>>>(W, Bp, K, Nc);
  };

  // one-time: pack weights into WMMA fragment order + encoders
  for (int l = 0; l < NLAYER; ++l) {
    packB(Wq  + l * WSZ, wq16 + l * WSZ, HIDDEN, HIDDEN);
    packB(Wk  + l * WSZ, wk16 + l * WSZ, HIDDEN, HIDDEN);
    packB(Wv  + l * WSZ, wv16 + l * WSZ, HIDDEN, HIDDEN);
    packB(We  + l * WSZ, we16 + l * WSZ, HIDDEN, HIDDEN);
    packB(Wsk + l * WSZ, ws16 + l * WSZ, HIDDEN, HIDDEN);
  }
  packB(W1, w116, HIDDEN, HIDDEN);
  packB(W2, w216, HIDDEN, NOUT);
  k_atom_encode<<<NNODE, HIDDEN, 0, stream>>>(atom_table, x, hf, hh);
  k_bond_encode<<<NEDGE, HIDDEN, 0, stream>>>(bond_table, edge_attr, eh);

  for (int l = 0; l < NLAYER; ++l) {
    launch_gemm<true,  false, true,  false>(hh, wq16 + l * WSZ, bq  + l * HIDDEN, qf,  nullptr, NNODE, HIDDEN, stream);
    launch_gemm<true,  false, true,  false>(hh, wk16 + l * WSZ, bk  + l * HIDDEN, kf,  nullptr, NNODE, HIDDEN, stream);
    launch_gemm<true,  false, true,  false>(hh, wv16 + l * WSZ, bvp + l * HIDDEN, vf,  nullptr, NNODE, HIDDEN, stream);
    launch_gemm<true,  false, true,  false>(hh, ws16 + l * WSZ, bsp + l * HIDDEN, skf, nullptr, NNODE, HIDDEN, stream);
    launch_gemm<false, false, false, true >(eh, we16 + l * WSZ, nullptr, nullptr, ee, NEDGE, HIDDEN, stream);

    k_fill<<<(NNODE * NHEAD + 255) / 256, 256, 0, stream>>>(nmax, -1e30f, NNODE * NHEAD);
    k_fill<<<(NNODE * NHEAD + 255) / 256, 256, 0, stream>>>(nsum, 0.f, NNODE * NHEAD);
    k_fill<<<(NNODE * HIDDEN + 255) / 256, 256, 0, stream>>>(msg, 0.f, NNODE * HIDDEN);

    k_edge_logits<<<(NEDGE + 7) / 8, 256, 0, stream>>>(edge_index, qf, kf, ee, lg, nmax);
    k_edge_exp<<<(NEDGE * NHEAD + 255) / 256, 256, 0, stream>>>(edge_index, lg, nmax, nsum);
    k_edge_accum<<<(NEDGE + 7) / 8, 256, 0, stream>>>(edge_index, lg, nsum, vf, ee, msg);
    k_node_update<<<(NNODE * HIDDEN + 255) / 256, 256, 0, stream>>>(msg, skf, hf, hh);
  }

  k_fill<<<(NGRAPH * HIDDEN + 255) / 256, 256, 0, stream>>>(gacc, 0.f, NGRAPH * HIDDEN);
  k_fill<<<(NGRAPH + 255) / 256, 256, 0, stream>>>(cnt, 0.f, NGRAPH);
  k_pool<<<NNODE, HIDDEN, 0, stream>>>(batch, hf, gacc, cnt);
  k_gdiv<<<NGRAPH, HIDDEN, 0, stream>>>(gacc, cnt, g16);

  launch_gemm<true, true,  false, true >(g16,  w116, b1, nullptr, g116, NGRAPH, HIDDEN, stream);
  launch_gemm<true, false, true,  false>(g116, w216, b2, out, nullptr, NGRAPH, NOUT, stream);
  k_l2norm<<<NGRAPH, NOUT, 0, stream>>>(out);
}
